// AttentionLayer_83004537963214
// MI455X (gfx1250) — matrix operand order repaired and checked by reading, not compiled, not run on hardware
//
#include <hip/hip_runtime.h>
#include <stddef.h>

// ---------------------------------------------------------------------------
// Types
// ---------------------------------------------------------------------------
typedef __bf16        v16bf __attribute__((ext_vector_type(16)));
typedef float         v8f   __attribute__((ext_vector_type(8)));
typedef unsigned int  u32x4 __attribute__((ext_vector_type(4)));

union FragBF {            // 16 bf16 operand for v_wmma_f32_16x16x32_bf16
    v16bf v;
    u32x4 q[2];
};

__device__ __forceinline__ unsigned short f32_to_bf16(float f) {
    unsigned int u = __float_as_uint(f);
    unsigned int rounding = 0x7FFFu + ((u >> 16) & 1u);   // round-to-nearest-even
    return (unsigned short)((u + rounding) >> 16);
}

// Generic pointer to LDS -> 32-bit group-segment offset (low 32 bits of the
// SHARED aperture address are the LDS offset).
__device__ __forceinline__ unsigned lds_off(const void* p) {
    return (unsigned)(unsigned long long)p;
}

// CDNA5 async copy: 16 bytes global -> LDS, tracked with ASYNCcnt.
__device__ __forceinline__ void async_cp16(unsigned ldsdst, const void* gsrc) {
    asm volatile("global_load_async_to_lds_b128 %0, %1, off"
                 :: "v"(ldsdst), "v"(gsrc) : "memory");
}
__device__ __forceinline__ void wait_async0() {
    asm volatile("s_wait_asynccnt 0x0" ::: "memory");
}

// ---------------------------------------------------------------------------
// fp32 -> bf16 conversions
// ---------------------------------------------------------------------------
__global__ __launch_bounds__(256)
void cvt_f32_bf16(const float* __restrict__ src, unsigned short* __restrict__ dst, size_t n) {
    size_t i = (size_t)blockIdx.x * blockDim.x + threadIdx.x;
    size_t stride = (size_t)gridDim.x * blockDim.x;
    for (; i < n; i += stride) dst[i] = f32_to_bf16(src[i]);
}

// Convert fp32 [K,N] row-major into the WMMA B-operand swizzled bf16 layout:
//   dst[ (kb*(N/16) + ng)*512 + ((k&16) + (n&15))*16 + (k&15) ]
// so a 32x128 B tile is one contiguous 8KB block (pure b128 copies in the GEMM).
__global__ __launch_bounds__(256)
void cvt_f32_bf16_swz(const float* __restrict__ src, unsigned short* __restrict__ dst,
                      int K, int N) {
    size_t total = (size_t)K * N;
    size_t i = (size_t)blockIdx.x * blockDim.x + threadIdx.x;
    size_t stride = (size_t)gridDim.x * blockDim.x;
    size_t ngrp = (size_t)(N >> 4);
    for (; i < total; i += stride) {
        int j = (int)(i & 15);              // k low bits
        int l = (int)((i >> 4) & 31);       // lane
        size_t rest = i >> 9;
        int ng = (int)(rest % ngrp);
        int kb = (int)(rest / ngrp);
        int k = kb * 32 + (l & 16) + j;
        int n = ng * 16 + (l & 15);
        dst[i] = f32_to_bf16(src[(size_t)k * N + n]);
    }
}

// ---------------------------------------------------------------------------
// Batched BF16 WMMA GEMM:  C[b] = act(A[b] @ B[b] + bias)
//   A : [M,K] bf16 row-major, batch stride sA (elems)
//   B : transB==0 -> pre-swizzled B-operand layout (see cvt_f32_bf16_swz)
//       transB==1 -> [N,K] row-major (logical B^T), staged contiguously
//   out32 / out16 : [M,N] fp32 / bf16 (batch stride sC); either may be null.
//   swz16 : write out16 in the swizzled B-operand layout (per batch of M rows)
// Block = 256 threads (8 wave32), tile 128x128, K-step 32, LDS double-buffered
// with global_load_async_to_lds_b128 prefetch of tile k+1 under the WMMAs of k.
// ---------------------------------------------------------------------------
__global__ __launch_bounds__(256)
void gemm_bf16_wmma(const unsigned short* __restrict__ A,
                    const unsigned short* __restrict__ B,
                    const float* __restrict__ bias,
                    float* __restrict__ out32,
                    unsigned short* __restrict__ out16,
                    int M, int N, int K,
                    long long sA, long long sB, long long sC,
                    int transB, int leaky, int swz16)
{
    __shared__ __align__(16) unsigned short lA[2][128 * 32];    // 2 x 8 KB
    __shared__ __align__(16) unsigned short lB[2][8 * 32 * 16]; // 2 x 8 KB

    const int tid  = threadIdx.x;
    const int wave = tid >> 5;
    const int lane = tid & 31;
    const int bm = blockIdx.y * 128;
    const int bn = blockIdx.x * 128;
    const int bz = blockIdx.z;

    const unsigned short* __restrict__ Ab = A + (long long)bz * sA;
    const unsigned short* __restrict__ Bb = B + (long long)bz * sB;

    // wave layout: 2 (M) x 4 (N) -> each wave owns 64 rows x 32 cols
    const int wm = (wave >> 2) * 64;
    const int wn = (wave & 3) * 32;

    v8f acc[4][2];
    {
        v8f z;
        for (int t = 0; t < 8; ++t) z[t] = 0.0f;
        for (int i = 0; i < 4; ++i)
            for (int j = 0; j < 2; ++j) acc[i][j] = z;
    }

    // async staging of one (A,B) K-tile into buffer `buf`
    auto stage = [&](int buf, int k0) {
        {   // A: 128 rows x 32 k; thread -> 32 contiguous bytes
            int row = tid >> 1;
            int seg = (tid & 1) * 16;
            const unsigned short* g = Ab + (size_t)(bm + row) * K + k0 + seg;
            unsigned d = lds_off(&lA[buf][row * 32 + seg]);
            async_cp16(d, g);
            async_cp16(d + 16, g + 8);
        }
        if (transB) {
            // B^T stored [N,K]: 16 consecutive k per thread -> one lane chunk
            int nl   = tid >> 1;
            int kseg = (tid & 1) * 16;
            const unsigned short* g = Bb + (size_t)(bn + nl) * K + k0 + kseg;
            int nt = nl >> 4;
            int ln = (nl & 15) + kseg;
            unsigned d = lds_off(&lB[buf][((nt << 5) + ln) * 16]);
            async_cp16(d, g);
            async_cp16(d + 16, g + 8);
        } else {
            // pre-swizzled global: contiguous 8 KB block per (k0, bn) tile
            const unsigned short* g =
                Bb + ((((size_t)(k0 >> 5) * (size_t)(N >> 4)) + (size_t)(bn >> 4)) << 9)
                   + (size_t)tid * 16;
            unsigned d = lds_off(&lB[buf][tid * 16]);
            async_cp16(d, g);
            async_cp16(d + 16, g + 8);
        }
    };

    stage(0, 0);
    wait_async0();
    __syncthreads();

    int buf = 0;
    for (int k0 = 0; k0 < K; k0 += 32, buf ^= 1) {
        if (k0 + 32 < K) stage(buf ^ 1, k0 + 32);   // prefetch under compute

        // ---- fragments -----------------------------------------------------
        // A (16x32 bf16): lanes 0-15 -> M=m0+lane, K=0..15
        //                 lanes16-31 -> M=m0+lane-16, K=16..31
        FragBF a[4];
        {
            int arow = wm + (lane & 15);
            int akb  = (lane < 16) ? 0 : 16;
            for (int i = 0; i < 4; ++i) {
                const unsigned short* p = &lA[buf][(arow + i * 16) * 32 + akb];
                a[i].q[0] = *(const u32x4*)(p);
                a[i].q[1] = *(const u32x4*)(p + 8);
            }
        }
        FragBF b[2];
        {
            int ntb = wn >> 4;
            for (int j = 0; j < 2; ++j) {
                const unsigned short* p = &lB[buf][(((ntb + j) << 5) + lane) * 16];
                b[j].q[0] = *(const u32x4*)(p);
                b[j].q[1] = *(const u32x4*)(p + 8);
            }
        }

        // ---- 8 WMMAs per K-step -------------------------------------------
        for (int i = 0; i < 4; ++i)
            for (int j = 0; j < 2; ++j)
                acc[i][j] = __builtin_amdgcn_wmma_f32_16x16x32_bf16(
                    false, a[i].v, false, b[j].v, (short)0, acc[i][j], false, false);

        wait_async0();      // prefetch landed
        __syncthreads();    // everyone done reading `buf`
    }

    // ---- epilogue: bias + optional LeakyReLU, fp32/bf16 stores -------------
    // C layout: VGPR r, lanes 0-15 -> M=r, N=lane; lanes 16-31 -> M=r+8
    const int rbase  = (lane < 16) ? 0 : 8;
    const int n_lane = lane & 15;
    const size_t cbase = (size_t)((long long)bz * sC);
    for (int i = 0; i < 4; ++i) {
        for (int j = 0; j < 2; ++j) {
            int col = bn + wn + j * 16 + n_lane;
            float bv = bias ? bias[col] : 0.0f;
            for (int r = 0; r < 8; ++r) {
                int row = bm + wm + i * 16 + rbase + r;
                float v = acc[i][j][r] + bv;
                if (leaky) v = (v > 0.0f) ? v : 0.01f * v;
                size_t idx = cbase + (size_t)row * N + col;
                if (out32) out32[idx] = v;
                if (out16) {
                    size_t i16;
                    if (swz16) {
                        i16 = cbase
                            + ((((size_t)(row >> 5) * (size_t)(N >> 4) + (size_t)(col >> 4)) << 9)
                               + (size_t)(((row & 16) + (col & 15)) << 4)
                               + (size_t)(row & 15));
                    } else {
                        i16 = idx;
                    }
                    out16[i16] = f32_to_bf16(v);
                }
            }
        }
    }
}

// ---------------------------------------------------------------------------
// Row softmax: one block per row (fp32 in, bf16 out).
// ---------------------------------------------------------------------------
__global__ __launch_bounds__(256)
void softmax_rows(const float* __restrict__ S, unsigned short* __restrict__ P, int cols) {
    __shared__ float red[9];
    const size_t row = blockIdx.x;
    const int tid = threadIdx.x;
    const int per = cols >> 8;
    const float* __restrict__ src = S + row * (size_t)cols;
    unsigned short* __restrict__ dst = P + row * (size_t)cols;

    float vals[8];
    float m = -3.4e38f;
    for (int i = 0; i < per; ++i) {
        vals[i] = src[tid + (i << 8)];
        m = fmaxf(m, vals[i]);
    }
    for (int o = 16; o > 0; o >>= 1) m = fmaxf(m, __shfl_xor(m, o, 32));
    if ((tid & 31) == 0) red[tid >> 5] = m;
    __syncthreads();
    if (tid == 0) {
        float t = red[0];
        for (int i = 1; i < 8; ++i) t = fmaxf(t, red[i]);
        red[8] = t;
    }
    __syncthreads();
    m = red[8];
    __syncthreads();

    float s = 0.0f;
    for (int i = 0; i < per; ++i) {
        vals[i] = __expf(vals[i] - m);
        s += vals[i];
    }
    for (int o = 16; o > 0; o >>= 1) s += __shfl_xor(s, o, 32);
    if ((tid & 31) == 0) red[tid >> 5] = s;
    __syncthreads();
    if (tid == 0) {
        float t = 0.0f;
        for (int i = 0; i < 8; ++i) t += red[i];
        red[8] = t;
    }
    __syncthreads();
    const float inv = 1.0f / red[8];
    for (int i = 0; i < per; ++i)
        dst[tid + (i << 8)] = f32_to_bf16(vals[i] * inv);
}

// ---------------------------------------------------------------------------
// Fused (residual +) LeakyReLU + LayerNorm. One block per row.
// ---------------------------------------------------------------------------
__global__ __launch_bounds__(256)
void ln_rows(const float* __restrict__ A, const float* __restrict__ Res,
             const float* __restrict__ g, const float* __restrict__ be,
             float* __restrict__ out32, unsigned short* __restrict__ out16,
             int cols, int leaky)
{
    __shared__ float red[9];
    const size_t row = blockIdx.x;
    const int tid = threadIdx.x;
    const int per = cols >> 8;
    const float* __restrict__ a = A + row * (size_t)cols;
    const float* __restrict__ r = Res ? Res + row * (size_t)cols : nullptr;

    float v[8];
    float s = 0.0f;
    for (int i = 0; i < per; ++i) {
        float t = a[tid + (i << 8)];
        if (r) t += r[tid + (i << 8)];
        if (leaky) t = (t > 0.0f) ? t : 0.01f * t;
        v[i] = t;
        s += t;
    }
    for (int o = 16; o > 0; o >>= 1) s += __shfl_xor(s, o, 32);
    if ((tid & 31) == 0) red[tid >> 5] = s;
    __syncthreads();
    if (tid == 0) {
        float t = 0.0f;
        for (int i = 0; i < 8; ++i) t += red[i];
        red[8] = t;
    }
    __syncthreads();
    const float mean = red[8] / (float)cols;
    __syncthreads();

    float q = 0.0f;
    for (int i = 0; i < per; ++i) {
        float d = v[i] - mean;
        q += d * d;
    }
    for (int o = 16; o > 0; o >>= 1) q += __shfl_xor(q, o, 32);
    if ((tid & 31) == 0) red[tid >> 5] = q;
    __syncthreads();
    if (tid == 0) {
        float t = 0.0f;
        for (int i = 0; i < 8; ++i) t += red[i];
        red[8] = t;
    }
    __syncthreads();
    const float rstd = rsqrtf(red[8] / (float)cols + 1e-5f);

    for (int i = 0; i < per; ++i) {
        int c = tid + (i << 8);
        float o = (v[i] - mean) * rstd * g[c] + be[c];
        size_t idx = row * (size_t)cols + c;
        if (out32) out32[idx] = o;
        if (out16) out16[idx] = f32_to_bf16(o);
    }
}

// ---------------------------------------------------------------------------
// Orchestration
// ---------------------------------------------------------------------------
extern "C" void kernel_launch(void* const* d_in, const int* in_sizes, int n_in,
                              void* d_out, int out_size, void* d_ws, size_t ws_size,
                              hipStream_t stream)
{
    const int  Bn = 8, Sq = 2048, D = 1024, H = 2048;
    const size_t BS = (size_t)Bn * Sq;

    const float* x    = (const float*)d_in[0];
    const float* wq   = (const float*)d_in[1];
    const float* bq   = (const float*)d_in[2];
    const float* wk   = (const float*)d_in[3];
    const float* bk   = (const float*)d_in[4];
    const float* wv   = (const float*)d_in[5];
    const float* bv   = (const float*)d_in[6];
    const float* w0   = (const float*)d_in[7];
    const float* b0   = (const float*)d_in[8];
    const float* w1   = (const float*)d_in[9];
    const float* b1   = (const float*)d_in[10];
    const float* ln_g = (const float*)d_in[11];
    const float* ln_b = (const float*)d_in[12];
    const float* w2   = (const float*)d_in[13];
    const float* b2   = (const float*)d_in[14];
    const float* n1_g = (const float*)d_in[15];
    const float* n1_b = (const float*)d_in[16];
    const float* n2_g = (const float*)d_in[17];
    const float* n2_b = (const float*)d_in[18];

    char* ws = (char*)d_ws;
    size_t off = 0;
    auto take = [&](size_t bytes) -> size_t {
        size_t r = off;
        off += (bytes + 255) & ~(size_t)255;
        return r;
    };

    // bf16 weights (swizzled B-operand layout)
    size_t o_wqb = take((size_t)D * D * 2);
    size_t o_wkb = take((size_t)D * D * 2);
    size_t o_wvb = take((size_t)D * D * 2);
    size_t o_w0b = take((size_t)D * D * 2);
    size_t o_w1b = take((size_t)D * H * 2);
    size_t o_w2b = take((size_t)H * D * 2);
    // activations (with buffer reuse)
    size_t o_xb  = take(BS * D * 2);                 // x bf16      -> later ln1 bf16
    size_t o_qb  = take(BS * D * 2);                 // q bf16      -> later e fp32 (with kb)
    size_t o_kb  = take(BS * D * 2);                 // k bf16         (contiguous after qb)
    size_t o_vb  = take(BS * D * 2);                 // v bf16 (swizzled) -> later e bf16
    size_t o_sc  = take((size_t)Bn * Sq * Sq * 4);   // scores fp32 -> later h fp32
    size_t o_pr  = take((size_t)Bn * Sq * Sq * 2);   // probs bf16  -> later logits fp32
    size_t o_att = take(BS * D * 4);                 // attn fp32   -> later h_ln bf16

    unsigned short* wqb  = (unsigned short*)(ws + o_wqb);
    unsigned short* wkb  = (unsigned short*)(ws + o_wkb);
    unsigned short* wvb  = (unsigned short*)(ws + o_wvb);
    unsigned short* w0b  = (unsigned short*)(ws + o_w0b);
    unsigned short* w1b  = (unsigned short*)(ws + o_w1b);
    unsigned short* w2b  = (unsigned short*)(ws + o_w2b);
    unsigned short* xb   = (unsigned short*)(ws + o_xb);
    unsigned short* qb   = (unsigned short*)(ws + o_qb);
    unsigned short* kb   = (unsigned short*)(ws + o_kb);
    unsigned short* vb   = (unsigned short*)(ws + o_vb);
    float*          sc   = (float*)(ws + o_sc);
    unsigned short* pr   = (unsigned short*)(ws + o_pr);
    float*          att  = (float*)(ws + o_att);
    // reuse aliases
    unsigned short* ln1b = (unsigned short*)(ws + o_xb);   // LN1 output bf16
    float*          e32  = (float*)(ws + o_qb);            // e fp32 (qb+kb region)
    unsigned short* eb   = (unsigned short*)(ws + o_vb);   // e bf16
    float*          h32  = (float*)(ws + o_sc);            // h fp32
    unsigned short* hlnb = (unsigned short*)(ws + o_att);  // LN(h) bf16
    float*          lg   = (float*)(ws + o_pr);            // logits fp32
    float*          out  = (float*)d_out;

    dim3 blk(256);
    auto cvt = [&](const float* s, unsigned short* d, size_t n) {
        int g = (int)((n + 256 * 8 - 1) / (256 * 8));
        cvt_f32_bf16<<<g, blk, 0, stream>>>(s, d, n);
    };
    auto cvtswz = [&](const float* s, unsigned short* d, int K, int N) {
        size_t n = (size_t)K * N;
        int g = (int)((n + 256 * 8 - 1) / (256 * 8));
        cvt_f32_bf16_swz<<<g, blk, 0, stream>>>(s, d, K, N);
    };
    auto gemm = [&](const unsigned short* A, const unsigned short* B, const float* bias,
                    float* o32, unsigned short* o16, int M, int N, int K,
                    long long sA, long long sB, long long sC, int nb, int tB,
                    int lk, int swz) {
        dim3 grid(N / 128, M / 128, nb);
        gemm_bf16_wmma<<<grid, blk, 0, stream>>>(A, B, bias, o32, o16,
                                                 M, N, K, sA, sB, sC, tB, lk, swz);
    };

    const long long SD = (long long)Sq * D;
    const long long SS = (long long)Sq * Sq;

    // 1) precision drop: x row-major, weights into the swizzled B layout
    cvt(x, xb, BS * D);
    cvtswz(wq, wqb, D, D);
    cvtswz(wk, wkb, D, D);
    cvtswz(wv, wvb, D, D);
    cvtswz(w0, w0b, D, D);
    cvtswz(w1, w1b, D, H);
    cvtswz(w2, w2b, H, D);

    // 2) Q,K,V projections (batched so v can be emitted per-batch swizzled)
    gemm(xb, wqb, bq, nullptr, qb, Sq, D, D, SD, 0, SD, Bn, 0, 0, 0);
    gemm(xb, wkb, bk, nullptr, kb, Sq, D, D, SD, 0, SD, Bn, 0, 0, 0);
    gemm(xb, wvb, bv, nullptr, vb, Sq, D, D, SD, 0, SD, Bn, 0, 0, 1); // swizzled

    // 3) scores[b] = q[b] @ k[b]^T : B transposed path
    gemm(qb, kb, nullptr, sc, nullptr, Sq, Sq, D, SD, SD, SS, Bn, 1, 0, 0);

    // 4) row softmax -> bf16 probs
    softmax_rows<<<dim3((unsigned)(Bn * Sq)), blk, 0, stream>>>(sc, pr, Sq);

    // 5) attn[b] = probs[b] @ v[b]  (v already in swizzled B layout)
    gemm(pr, vb, nullptr, att, nullptr, Sq, D, Sq, SS, SD, SD, Bn, 0, 0, 0);

    // 6) ln1 = LN(x + attn) -> bf16
    ln_rows<<<dim3((unsigned)BS), blk, 0, stream>>>(att, x, n1_g, n1_b,
                                                    nullptr, ln1b, D, 0);

    // 7) e = ln1 @ w0 + b0 (fp32 for the final residual, bf16 for the MLP)
    gemm(ln1b, w0b, b0, e32, eb, (int)BS, D, D, 0, 0, 0, 1, 0, 0, 0);

    // 8) h = leaky_relu(e @ w1 + b1)
    gemm(eb, w1b, b1, h32, nullptr, (int)BS, H, D, 0, 0, 0, 1, 0, 1, 0);

    // 9) h = LN(h) -> bf16
    ln_rows<<<dim3((unsigned)BS), blk, 0, stream>>>(h32, nullptr, ln_g, ln_b,
                                                    nullptr, hlnb, H, 0);

    // 10) logits = h @ w2 + b2
    gemm(hlnb, w2b, b2, lg, nullptr, (int)BS, D, H, 0, 0, 0, 1, 0, 0, 0);

    // 11) out = LN(leaky_relu(logits + e))
    ln_rows<<<dim3((unsigned)BS), blk, 0, stream>>>(lg, e32, n2_g, n2_b,
                                                    out, nullptr, D, 1);

    (void)in_sizes; (void)n_in; (void)out_size; (void)ws_size;
}
